// auxiliary_y_fixed_15083925143707
// MI455X (gfx1250) — matrix operand order repaired and checked by reading, not compiled, not run on hardware
//
#include <hip/hip_runtime.h>

typedef _Float16 v4h  __attribute__((ext_vector_type(4)));
typedef _Float16 v8h  __attribute__((ext_vector_type(8)));
typedef _Float16 v16h __attribute__((ext_vector_type(16)));
typedef float    v8f  __attribute__((ext_vector_type(8)));

#define NTOT   262144
#define NBAGS  8192
#define DIN    128
#define DCAT   256
#define HID    512
#define TM     128          // rows per workgroup

// LDS row strides padded by 8 halves (16B) to de-conflict banked gathers:
// row-to-row delta = stride*2 bytes; /4 mod 64 banks == 4 -> conflict free.
#define AS_STRIDE  (DCAT + 8)   // 264 halves
#define HS_STRIDE  (HID + 8)    // 520 halves

// workspace layout (in _Float16 elements unless noted)
#define W1S_ELEMS   (DCAT*HID)            // 131072 halves, swizzled W1
#define W2S_ELEMS   (HID*HID)             // 262144 halves, swizzled W2
#define W2S_OFF     (W1S_ELEMS)
#define BAG_BYTE_OFF ((size_t)2*(W1S_ELEMS + W2S_ELEMS))   // 786432 bytes

// ---------------------------------------------------------------------------
// order-preserving float <-> uint mapping for atomic max
__device__ __forceinline__ unsigned fmap(float f) {
    unsigned b = __float_as_uint(f);
    return (b & 0x80000000u) ? ~b : (b | 0x80000000u);
}
__device__ __forceinline__ float funmap(unsigned u) {
    unsigned b = (u & 0x80000000u) ? (u ^ 0x80000000u) : ~u;
    return __uint_as_float(b);
}

// Gather a 16x32 f16 A-fragment (ISA layout) from row-major LDS tile.
// lanes 0-15:  elems 0..7 -> K c0..c0+7,  elems 8..15 -> K c0+16..c0+23 (c0 = 32*kc)
// lanes 16-31: same with c0 += 8
__device__ __forceinline__ v16h gather_a(const _Float16* base, int stride,
                                         int lane, int mrow0, int kc) {
    int m  = mrow0 + (lane & 15);
    int c0 = (kc << 5) + ((lane >> 4) << 3);
    const _Float16* p = base + m * stride + c0;
    union { v16h v; v8h h[2]; } u;
    u.h[0] = *(const v8h*)(p);
    u.h[1] = *(const v8h*)(p + 16);
    return u.v;
}

// ---------------------------------------------------------------------------
// Prep: convert W1/W2 fp32 -> f16, pre-swizzled into WMMA B-fragment order:
//   idx = (((ntile*KC + kc)*32) + lane)*16 + e
//   k   = kc*32 + (e>=8 ? 16 : 0) + (lane>=16 ? 8 : 0) + (e & 7) ;  n = ntile*16 + (lane & 15)
// Also zero the per-bag max buffer.
__global__ void prep_kernel(const float* __restrict__ W1,
                            const float* __restrict__ W2,
                            _Float16* __restrict__ wsh,
                            unsigned* __restrict__ bagbuf) {
    int id = blockIdx.x * 256 + threadIdx.x;
    if (id < W1S_ELEMS) {
        int e    = id & 15;
        int lane = (id >> 4) & 31;
        int rest = id >> 9;
        int kc   = rest & 7;        // K=256 -> 8 k-chunks
        int nt   = rest >> 3;
        int k    = (kc << 5) + ((e >> 3) << 4) + ((lane >> 4) << 3) + (e & 7);
        int n    = (nt << 4) + (lane & 15);
        wsh[id] = (_Float16)W1[k * HID + n];
    } else if (id < W1S_ELEMS + W2S_ELEMS) {
        int j    = id - W1S_ELEMS;
        int e    = j & 15;
        int lane = (j >> 4) & 31;
        int rest = j >> 9;
        int kc   = rest & 15;       // K=512 -> 16 k-chunks
        int nt   = rest >> 4;
        int k    = (kc << 5) + ((e >> 3) << 4) + ((lane >> 4) << 3) + (e & 7);
        int n    = (nt << 4) + (lane & 15);
        wsh[W2S_OFF + j] = (_Float16)W2[k * HID + n];
    } else if (id < W1S_ELEMS + W2S_ELEMS + NBAGS) {
        bagbuf[id - (W1S_ELEMS + W2S_ELEMS)] = 0u;   // < fmap(any finite)
    }
}

// ---------------------------------------------------------------------------
// Fused MLP: per workgroup, 128 rows through concat->512->512->1, then
// per-bag atomic max.
// 8 waves = 2 M-groups (64 rows) x 4 N-groups (128 cols).  Each wave processes
// its 8 N-tiles in 2 chunks of 4, so each A-fragment load feeds 16 WMMAs
// -> LDS gather traffic sits at the 256B/cycle/WGP balance point.
__global__ __launch_bounds__(256)
void mil_mlp_kernel(const float* __restrict__ z_ins, const float* __restrict__ z_bag,
                    const int* __restrict__ bag_idx,
                    const float* __restrict__ b1, const float* __restrict__ b2,
                    const float* __restrict__ W3, const float* __restrict__ b3,
                    const _Float16* __restrict__ wsh,
                    unsigned* __restrict__ bagbuf,
                    float* __restrict__ out) {
    extern __shared__ _Float16 smem[];
    _Float16* As   = smem;                                   // [128][264] f16
    _Float16* Hs   = smem + TM * AS_STRIDE;                  // [128][520] f16
    float*    locp = (float*)(smem + TM * AS_STRIDE + TM * HS_STRIDE); // [128]

    const int tid  = threadIdx.x;
    const int lane = tid & 31;
    const int wv   = tid >> 5;
    const int mgrp = wv >> 2;       // 0..1 : rows [mgrp*64, +64)
    const int ngrp = wv & 3;        // 0..3 : cols [ngrp*128, +128)
    const int row0 = blockIdx.x * TM;

    if (tid < TM) locp[tid] = 0.0f;

    // ---- stage inputs: concat(z_ins, z_bag) fp32 -> f16 in LDS ----
    for (int s = tid; s < TM * (DCAT / 4); s += 256) {
        int r  = s >> 6;                        // DCAT/4 == 64
        int c4 = s & 63;
        const float* src = (c4 < 32)
            ? (z_ins + (size_t)(row0 + r) * DIN + (c4 << 2))
            : (z_bag + (size_t)(row0 + r) * DIN + ((c4 - 32) << 2));
        float4 f = *(const float4*)src;
        v4h h = { (_Float16)f.x, (_Float16)f.y, (_Float16)f.z, (_Float16)f.w };
        *(v4h*)(As + r * AS_STRIDE + (c4 << 2)) = h;
    }
    __syncthreads();

    // ---- GEMM1: Hs = relu(As[128x256] * W1[256x512] + b1) ----
    #pragma unroll
    for (int ch = 0; ch < 2; ++ch) {
        v8f acc[4][4] = {};
        #pragma unroll 1
        for (int kc = 0; kc < 8; ++kc) {
            v16h Af[4];
            #pragma unroll
            for (int mt = 0; mt < 4; ++mt)
                Af[mt] = gather_a(As, AS_STRIDE, lane, (mgrp << 6) + (mt << 4), kc);
            #pragma unroll
            for (int ntl = 0; ntl < 4; ++ntl) {
                const int ntile = (ngrp << 3) + (ch << 2) + ntl;
                const _Float16* bp = wsh + (size_t)((((ntile << 3) + kc) << 5) + lane) * 16;
                v16h B = *(const v16h*)bp;
                __builtin_prefetch(bp + 32 * 16, 0, 3);   // next kc, WGP-scope (near)
                #pragma unroll
                for (int mt = 0; mt < 4; ++mt)
                    acc[ntl][mt] = __builtin_amdgcn_wmma_f32_16x16x32_f16(
                        false, Af[mt], false, B, (short)0, acc[ntl][mt], false, false);
            }
        }
        const int mo = (lane >> 4) << 3;
        #pragma unroll
        for (int ntl = 0; ntl < 4; ++ntl) {
            const int   n  = (((ngrp << 3) + (ch << 2) + ntl) << 4) + (lane & 15);
            const float bv = b1[n];
            #pragma unroll
            for (int mt = 0; mt < 4; ++mt)
                #pragma unroll
                for (int v = 0; v < 8; ++v) {
                    float x = acc[ntl][mt][v] + bv;
                    Hs[((mgrp << 6) + (mt << 4) + mo + v) * HS_STRIDE + n] =
                        (_Float16)(x > 0.0f ? x : 0.0f);
                }
        }
    }
    __syncthreads();

    // ---- GEMM2 (+fused GEMM3): h2 = relu(Hs*W2 + b2); locp += h2 .* W3 ----
    const _Float16* W2s = wsh + W2S_OFF;
    #pragma unroll
    for (int ch = 0; ch < 2; ++ch) {
        v8f acc[4][4] = {};
        #pragma unroll 1
        for (int kc = 0; kc < 16; ++kc) {
            v16h Af[4];
            #pragma unroll
            for (int mt = 0; mt < 4; ++mt)
                Af[mt] = gather_a(Hs, HS_STRIDE, lane, (mgrp << 6) + (mt << 4), kc);
            #pragma unroll
            for (int ntl = 0; ntl < 4; ++ntl) {
                const int ntile = (ngrp << 3) + (ch << 2) + ntl;
                const _Float16* bp = W2s + (size_t)((((ntile << 4) + kc) << 5) + lane) * 16;
                v16h B = *(const v16h*)bp;
                __builtin_prefetch(bp + 32 * 16, 0, 3);   // next kc, WGP-scope (near)
                #pragma unroll
                for (int mt = 0; mt < 4; ++mt)
                    acc[ntl][mt] = __builtin_amdgcn_wmma_f32_16x16x32_f16(
                        false, Af[mt], false, B, (short)0, acc[ntl][mt], false, false);
            }
        }
        const int mo = (lane >> 4) << 3;
        #pragma unroll
        for (int ntl = 0; ntl < 4; ++ntl) {
            const int   n   = (((ngrp << 3) + (ch << 2) + ntl) << 4) + (lane & 15);
            const float bv  = b2[n];
            const float w3v = W3[n];
            #pragma unroll
            for (int mt = 0; mt < 4; ++mt)
                #pragma unroll
                for (int v = 0; v < 8; ++v) {
                    float x = acc[ntl][mt][v] + bv;
                    x = x > 0.0f ? x : 0.0f;
                    atomicAdd(&locp[(mgrp << 6) + (mt << 4) + mo + v], x * w3v);
                }
        }
    }
    __syncthreads();

    // ---- write loc, per-bag atomic max ----
    if (tid < TM) {
        const int   row = row0 + tid;
        const float loc = locp[tid] + b3[0];
        out[(size_t)3 * NTOT + row] = loc;
        atomicMax(&bagbuf[bag_idx[row]], fmap(loc));
    }
}

// ---------------------------------------------------------------------------
// Finalize: M[i] = seg_max(bag i) for i < NBAGS else 0; write three copies.
__global__ void finalize_kernel(const unsigned* __restrict__ bagbuf,
                                float* __restrict__ out) {
    int i = blockIdx.x * 256 + threadIdx.x;
    if (i >= NTOT) return;
    float v = 0.0f;
    if (i < NBAGS) {
        unsigned u = bagbuf[i];
        if (u != 0u) v = funmap(u);             // empty bags stay 0
    }
    out[i]            = v;
    out[NTOT + i]     = v;
    out[2 * NTOT + i] = v;
}

// ---------------------------------------------------------------------------
extern "C" void kernel_launch(void* const* d_in, const int* in_sizes, int n_in,
                              void* d_out, int out_size, void* d_ws, size_t ws_size,
                              hipStream_t stream) {
    const float* z_ins   = (const float*)d_in[0];
    const float* z_bag   = (const float*)d_in[1];
    const int*   bag_idx = (const int*)d_in[2];
    // d_in[3] = bag_latent_embeddings (unused by reference)
    const float* W1 = (const float*)d_in[4];
    const float* b1 = (const float*)d_in[5];
    const float* W2 = (const float*)d_in[6];
    const float* b2 = (const float*)d_in[7];
    const float* W3 = (const float*)d_in[8];
    const float* b3 = (const float*)d_in[9];
    float* out = (float*)d_out;

    _Float16* wsh    = (_Float16*)d_ws;
    unsigned* bagbuf = (unsigned*)((char*)d_ws + BAG_BYTE_OFF);

    const int prep_items = W1S_ELEMS + W2S_ELEMS + NBAGS;
    prep_kernel<<<(prep_items + 255) / 256, 256, 0, stream>>>(W1, W2, wsh, bagbuf);

    const size_t smem_bytes = (size_t)TM * AS_STRIDE * sizeof(_Float16)
                            + (size_t)TM * HS_STRIDE * sizeof(_Float16)
                            + (size_t)TM * sizeof(float);   // 201,216 B (< 320KB/WGP)
    mil_mlp_kernel<<<NTOT / TM, 256, smem_bytes, stream>>>(
        z_ins, z_bag, bag_idx, b1, b2, W3, b3, wsh, bagbuf, out);

    finalize_kernel<<<(NTOT + 255) / 256, 256, 0, stream>>>(bagbuf, out);
}
